// TextLSTM_36258113912845
// MI455X (gfx1250) — compile-verified, matrix operand order checked by reading, not run on hardware
//
#include <hip/hip_runtime.h>
#include <hip/hip_bf16.h>

// Problem sizes (fixed by the reference)
#define B_      256
#define T_      512
#define EMB_    128
#define HID_    256
#define NCLS_   32000

typedef __attribute__((ext_vector_type(16))) _Float16 v16h;
typedef __attribute__((ext_vector_type(8)))  _Float16 v8h;
typedef __attribute__((ext_vector_type(4)))  _Float16 v4h;
typedef __attribute__((ext_vector_type(8)))  float    v8f;
typedef int v4i __attribute__((vector_size(16)));   // matches async-LDS builtin param

__device__ __forceinline__ float sigmoidf_(float x) {
    return 1.0f / (1.0f + __expf(-x));
}

// ---------------------------------------------------------------------------
// Kernel 1: layer-0 LSTM gates at the LAST timestep only (layer-0 carry is
// provably zero in the reference; the layer-1 scan is dead code w.r.t. output).
// h0[b,h] = sigmoid(E·U_o+b_o) * tanh( sigmoid(E·U_i+b_i) * tanh(E·U_c+b_c) )
// ---------------------------------------------------------------------------
__global__ __launch_bounds__(HID_)
void lstm_gates_last(const int* __restrict__ X,
                     const float* __restrict__ C_table,
                     const float* __restrict__ U_i, const float* __restrict__ b_i,
                     const float* __restrict__ U_c, const float* __restrict__ b_c,
                     const float* __restrict__ U_o, const float* __restrict__ b_o,
                     _Float16* __restrict__ h0f16) {
    __shared__ float sE[EMB_];
    const int b = blockIdx.x;
    const int h = threadIdx.x;

    if (h < EMB_) {
        const int tok = X[b * T_ + (T_ - 1)];
        sE[h] = C_table[(size_t)tok * EMB_ + h];
    }
    __syncthreads();

    float ai = b_i[h];
    float ac = b_c[h];
    float ao = b_o[h];
    #pragma unroll 4
    for (int e = 0; e < EMB_; ++e) {
        const float ev = sE[e];
        ai = fmaf(ev, U_i[e * HID_ + h], ai);   // coalesced across h
        ac = fmaf(ev, U_c[e * HID_ + h], ac);
        ao = fmaf(ev, U_o[e * HID_ + h], ao);
    }
    const float i0 = sigmoidf_(ai);
    const float g0 = tanhf(ac);
    const float o0 = sigmoidf_(ao);
    const float c0 = i0 * g0;                   // f-gate * c_prev == 0 exactly
    const float h0 = o0 * tanhf(c0);

    h0f16[b * HID_ + h] = (_Float16)h0;
}

// ---------------------------------------------------------------------------
// Kernel 2: logits[M=256, N=32000] = h0f16[M,K=256] * W_w[N,K]^T + b_out
// HBM-bound (~66 MB @ 23.3 TB/s ≈ 2.8 us floor): W_w read once, logits
// written once (nontemporal). Workgroup tile 64(M) x 128(N), full K resident
// for A; B double-buffered in 32-K chunks with register-pipelined staging.
// 8 waves, each owns a 32x32 tile -> 4 x v_wmma_f32_16x16x32_f16 per K-step.
// ---------------------------------------------------------------------------
#define SA_ROWS   64
#define SA_STRIDE 264   // 256 + 8 f16 pad (row = 528B, 16B aligned)
#define SB_ROWS   128
#define SB_STRIDE 40    // 32 + 8 f16 pad  (row = 80B, 16B aligned)
#define KCHUNK    32
#define NCHUNKS   (HID_ / KCHUNK)   // 8

#define SB_BUF    (SB_ROWS * SB_STRIDE)

#if defined(__gfx1250__) && __has_builtin(__builtin_amdgcn_global_load_async_to_lds_b128) && \
    __has_builtin(__builtin_amdgcn_s_wait_asynccnt)
#define USE_ASYNC_LDS 1
#else
#define USE_ASYNC_LDS 0
#endif

__device__ __forceinline__ v16h load_frag_lds(const _Float16* p) {
    // two contiguous 8-f16 (16B) chunks: K = base..base+7 and base+16..base+23
    v8h lo = *(const v8h*)(p);
    v8h hi = *(const v8h*)(p + 16);
    return __builtin_shufflevector(lo, hi, 0,1,2,3,4,5,6,7,8,9,10,11,12,13,14,15);
}

__device__ __forceinline__ void load_B_chunk(const float* __restrict__ Ww, int n0,
                                             int kc, int tid, float4 r[4]) {
    #pragma unroll
    for (int it = 0; it < 4; ++it) {
        const int idx = tid + it * 256;      // 1024 float4 groups
        const int row = idx >> 3;            // 0..127
        const int c4  = idx & 7;             // float4 within 32 floats
        r[it] = *(const float4*)(Ww + (size_t)(n0 + row) * HID_ + kc * KCHUNK + c4 * 4);
    }
}

__device__ __forceinline__ void store_B_chunk(_Float16* __restrict__ sBbuf, int tid,
                                              const float4 r[4]) {
    #pragma unroll
    for (int it = 0; it < 4; ++it) {
        const int idx = tid + it * 256;
        const int row = idx >> 3;
        const int c4  = idx & 7;
        v4h hv;
        hv[0] = (_Float16)r[it].x; hv[1] = (_Float16)r[it].y;
        hv[2] = (_Float16)r[it].z; hv[3] = (_Float16)r[it].w;
        *(v4h*)(&sBbuf[row * SB_STRIDE + c4 * 4]) = hv;     // ds_store_b64
    }
}

__global__ __launch_bounds__(256)
void logits_wmma(const _Float16* __restrict__ h0f16,
                 const float* __restrict__ Ww,
                 const float* __restrict__ b_out,
                 float* __restrict__ out) {
    __shared__ _Float16 sA[SA_ROWS * SA_STRIDE];   // ~33.8 KB
    __shared__ _Float16 sB[2 * SB_BUF];            // ~20.0 KB (double buffer)

    const int tid  = threadIdx.x;
    const int n0   = blockIdx.x * 128;
    const int m0   = blockIdx.y * 64;
    const int wid  = tid >> 5;        // 8 waves (wave32)
    const int lane = tid & 31;
    const int r16  = lane & 15;
    const int half = lane >> 4;
    const int m_wave = (wid >> 2) * 32;   // 2 M-groups
    const int n_wave = (wid & 3)  * 32;   // 4 N-groups

    // ---- stage the entire A tile (64 rows x K=256 f16) into LDS, once ----
#if USE_ASYNC_LDS
    #pragma unroll
    for (int it = 0; it < 8; ++it) {
        const int idx = tid + it * 256;          // 2048 x 16B
        const int r = idx >> 5;                  // 0..63
        const int c = idx & 31;                  // uint4 within row
        __builtin_amdgcn_global_load_async_to_lds_b128(
            (v4i*)(h0f16 + (m0 + r) * HID_ + c * 8),
            (v4i*)&sA[r * SA_STRIDE + c * 8],
            /*offset=*/0, /*cpol=*/0);
    }
#else
    #pragma unroll
    for (int it = 0; it < 8; ++it) {
        const int idx = tid + it * 256;
        const int r = idx >> 5;
        const int c = idx & 31;
        const uint4 v = *(const uint4*)(h0f16 + (m0 + r) * HID_ + c * 8);
        *(uint4*)(&sA[r * SA_STRIDE + c * 8]) = v;
    }
#endif

    // ---- pipeline prologue: B chunk 0 into registers ----
    float4 breg[4];
    load_B_chunk(Ww, n0, 0, tid, breg);

#if USE_ASYNC_LDS
    __builtin_amdgcn_s_wait_asynccnt(0);   // A-tile async copies complete (this wave)
#endif

    v8f acc[2][2];
    #pragma unroll
    for (int mt = 0; mt < 2; ++mt)
        #pragma unroll
        for (int nt = 0; nt < 2; ++nt)
            acc[mt][nt] = (v8f){0.f,0.f,0.f,0.f,0.f,0.f,0.f,0.f};

    #pragma unroll
    for (int kc = 0; kc < NCHUNKS; ++kc) {
        // convert & publish current chunk into buffer kc&1
        store_B_chunk(&sB[(kc & 1) * SB_BUF], tid, breg);
        __syncthreads();   // sB[kc&1] visible; also orders A-tile stage (kc==0)

        // start next chunk's global loads NOW -> latency hidden behind WMMA
        if (kc + 1 < NCHUNKS) load_B_chunk(Ww, n0, kc + 1, tid, breg);
        if (kc + 2 < NCHUNKS) {
            const int pr = tid >> 1;
            const int pc = (tid & 1) * 16;
            __builtin_prefetch(Ww + (size_t)(n0 + pr) * HID_ + (kc + 2) * KCHUNK + pc, 0, 1);
        }

        // ---- compute: one K=32 WMMA step over the wave's 32x32 tile ----
        const _Float16* sBcur = &sB[(kc & 1) * SB_BUF];
        v16h aF[2], bF[2];
        #pragma unroll
        for (int mt = 0; mt < 2; ++mt)
            aF[mt] = load_frag_lds(&sA[(m_wave + mt * 16 + r16) * SA_STRIDE
                                       + kc * KCHUNK + 8 * half]);
        #pragma unroll
        for (int nt = 0; nt < 2; ++nt)
            bF[nt] = load_frag_lds(&sBcur[(n_wave + nt * 16 + r16) * SB_STRIDE
                                          + 8 * half]);
        #pragma unroll
        for (int mt = 0; mt < 2; ++mt)
            #pragma unroll
            for (int nt = 0; nt < 2; ++nt)
                acc[mt][nt] = __builtin_amdgcn_wmma_f32_16x16x32_f16(
                    false, aF[mt], false, bF[nt],
                    (short)0, acc[mt][nt], false, false);
        // Note: writes at iteration kc+1 target buffer (kc+1)&1, whose last
        // readers finished at kc-1 (before this iteration's __syncthreads),
        // so one barrier per iteration is sufficient.
    }

    // ---- epilogue: C/D layout -> VGPR j holds row (j + 8*half), col r16 ----
    // logits are write-once / never-reread: stream out nontemporally.
    #pragma unroll
    for (int nt = 0; nt < 2; ++nt) {
        const int gn = n0 + n_wave + nt * 16 + r16;
        const float bias = b_out[gn];
        #pragma unroll
        for (int mt = 0; mt < 2; ++mt) {
            const int gmBase = m0 + m_wave + mt * 16 + 8 * half;
            #pragma unroll
            for (int j = 0; j < 8; ++j) {
                __builtin_nontemporal_store(acc[mt][nt][j] + bias,
                                            &out[(size_t)(gmBase + j) * NCLS_ + gn]);
            }
        }
    }
}

// ---------------------------------------------------------------------------
extern "C" void kernel_launch(void* const* d_in, const int* in_sizes, int n_in,
                              void* d_out, int out_size, void* d_ws, size_t ws_size,
                              hipStream_t stream) {
    const int*   X     = (const int*)  d_in[0];
    const float* Ct    = (const float*)d_in[1];
    const float* U_i   = (const float*)d_in[2];
    const float* b_i   = (const float*)d_in[4];
    const float* U_c   = (const float*)d_in[8];
    const float* b_c   = (const float*)d_in[10];
    const float* U_o   = (const float*)d_in[11];
    const float* b_o   = (const float*)d_in[13];
    const float* Ww    = (const float*)d_in[26];
    const float* bOut  = (const float*)d_in[27];

    _Float16* h0f16 = (_Float16*)d_ws;          // 256*256*2 = 128 KB scratch
    float*    out   = (float*)d_out;            // [256, 32000] f32

    lstm_gates_last<<<B_, HID_, 0, stream>>>(X, Ct, U_i, b_i, U_c, b_c, U_o, b_o, h0f16);

    dim3 grid(NCLS_ / 128, B_ / 64);            // (250, 4)
    logits_wmma<<<grid, 256, 0, stream>>>(h0f16, Ww, bOut, out);
}
